// ContagionGNN_26972394618971
// MI455X (gfx1250) — matrix operand order repaired and checked by reading, not compiled, not run on hardware
//
#include <hip/hip_runtime.h>
#include <hip/hip_bf16.h>

#define NN 100000       // nodes
#define NE 1600000      // edges
#define ND 128          // node dim
#define H  64           // hidden
#define OD 21           // out dim
#define SLOPE 0.2f

typedef __attribute__((ext_vector_type(16))) __bf16 bf16x16;
typedef __attribute__((ext_vector_type(8)))  float  f32x8;
typedef __attribute__((ext_vector_type(4)))  float  f32x4v;
typedef __attribute__((ext_vector_type(2)))  float  f32x2v;
typedef __attribute__((ext_vector_type(4)))  unsigned int u32x4v;

union FragCvt { uint4 q[2]; bf16x16 v; };

__device__ __forceinline__ unsigned short f2bf(float x) {
  union { float f; unsigned int u; } c; c.f = x;
  unsigned int r = c.u + 0x7FFFu + ((c.u >> 16) & 1u);   // round-to-nearest-even
  return (unsigned short)(r >> 16);
}
__device__ __forceinline__ float bf2f(unsigned short h) {
  union { unsigned int u; float f; } c; c.u = ((unsigned int)h) << 16; return c.f;
}
__device__ __forceinline__ float lrelu(float v) { return v > 0.f ? v : SLOPE * v; }

// A fragment: 16x32 bf16 tile from row-major LDS tile (stride S ushorts).
// lanes 0-15: M=lane, K = k0+{0..7} then k0+{16..23}
// lanes 16-31: M=lane-16, K = k0+{8..15} then k0+{24..31}
__device__ __forceinline__ bf16x16 load_a(const unsigned short* tile, int S, int lane, int k0) {
  int m  = lane & 15;
  int kb = k0 + ((lane >> 4) << 3);
  const unsigned short* p = tile + m * S + kb;
  FragCvt f;
  f.q[0] = *(const uint4*)(p);
  f.q[1] = *(const uint4*)(p + 16);
  return f.v;
}

// B fragment: 32x16 bf16, weights stored transposed in LDS as Wt[n][k] (stride S).
// lanes 0-15: N=lane, K = k0+{0..7} then k0+{16..23} ; lanes 16-31: N=lane-16, K = k0+{8..15} then k0+{24..31}
__device__ __forceinline__ bf16x16 load_b(const unsigned short* wt, int S, int lane, int ntile, int k0) {
  int n  = (ntile << 4) + (lane & 15);
  int kb = k0 + ((lane >> 4) << 3);
  const unsigned short* p = wt + n * S + kb;
  FragCvt f;
  f.q[0] = *(const uint4*)(p);
  f.q[1] = *(const uint4*)(p + 16);
  return f.v;
}

__device__ __forceinline__ f32x8 wmma_bf16(bf16x16 a, bf16x16 b, f32x8 c) {
  return __builtin_amdgcn_wmma_f32_16x16x32_bf16(false, a, false, b, (short)0, c, false, false);
}

#define WAVE_WAIT_DS() asm volatile("s_wait_dscnt 0x0" ::: "memory")

// ---------------------------------------------------------------------------
// e[E,64] (bf16) = lrelu(edge_attr[E,64] @ W[64,64] + b)
// edge_attr read + e write are pure streams -> non-temporal (protect L2)
// ---------------------------------------------------------------------------
__global__ void __launch_bounds__(256)
edge_linear_kernel(const float* __restrict__ ea, const float* __restrict__ W,
                   const float* __restrict__ bias, unsigned short* __restrict__ eout) {
  __shared__ __align__(16) unsigned short wt[64 * 64];       // Wt[n][k]
  __shared__ __align__(16) unsigned short astage[8][16 * 64];
  __shared__ float bsh[64];

  int tid = threadIdx.x;
  for (int i = tid; i < 64 * 64; i += 256) {
    int k = i >> 6, n = i & 63;
    wt[n * 64 + k] = f2bf(W[i]);
  }
  if (tid < 64) bsh[tid] = bias[tid];
  __syncthreads();

  int wave = tid >> 5, lane = tid & 31;
  long long row0 = (long long)blockIdx.x * 128 + wave * 16;
  unsigned short* at = astage[wave];

  // stage 16 rows x 64 cols, f32 -> bf16 (coalesced NT float2 per lane per row)
  #pragma unroll
  for (int m = 0; m < 16; ++m) {
    long long r = row0 + m;
    f32x2v v = {0.f, 0.f};
    if (r < NE) v = __builtin_nontemporal_load((const f32x2v*)(ea + r * 64 + lane * 2));
    *(unsigned int*)&at[m * 64 + lane * 2] =
        (unsigned int)f2bf(v.x) | ((unsigned int)f2bf(v.y) << 16);
  }
  WAVE_WAIT_DS();

  f32x8 acc[4] = {};
  #pragma unroll
  for (int k0 = 0; k0 < 64; k0 += 32) {
    bf16x16 a = load_a(at, 64, lane, k0);
    #pragma unroll
    for (int c = 0; c < 4; ++c) {
      bf16x16 b = load_b(wt, 64, lane, c, k0);
      acc[c] = wmma_bf16(a, b, acc[c]);
    }
  }

  // epilogue via LDS (bf16) for coalesced NT b32 stores
  int n = lane & 15, hi = lane >> 4;
  #pragma unroll
  for (int c = 0; c < 4; ++c) {
    int col = c * 16 + n;
    float bb = bsh[col];
    #pragma unroll
    for (int r = 0; r < 8; ++r)
      at[(hi * 8 + r) * 64 + col] = f2bf(lrelu(acc[c][r] + bb));
  }
  WAVE_WAIT_DS();
  #pragma unroll
  for (int m = 0; m < 16; ++m) {
    long long r = row0 + m;
    if (r < NE)
      __builtin_nontemporal_store(*(const unsigned int*)&at[m * 64 + lane * 2],
                                  (unsigned int*)(eout + r * 64 + lane * 2));
  }
}

// ---------------------------------------------------------------------------
// h[N,64] (f32) = lrelu(x[N,128] @ W[128,64] + b)
// x is streamed once -> NT; h is the gather target -> keep cacheable (RT)
// ---------------------------------------------------------------------------
__global__ void __launch_bounds__(256)
node_linear_kernel(const float* __restrict__ x, const float* __restrict__ W,
                   const float* __restrict__ bias, float* __restrict__ h) {
  __shared__ __align__(16) unsigned short wt[64 * 128];       // Wt[n][k], k<128
  __shared__ __align__(16) unsigned short astage[8][16 * 128];
  __shared__ float bsh[64];

  int tid = threadIdx.x;
  for (int i = tid; i < 128 * 64; i += 256) {
    int k = i >> 6, n = i & 63;
    wt[n * 128 + k] = f2bf(W[i]);
  }
  if (tid < 64) bsh[tid] = bias[tid];
  __syncthreads();

  int wave = tid >> 5, lane = tid & 31;
  long long row0 = (long long)blockIdx.x * 128 + wave * 16;
  unsigned short* at = astage[wave];

  #pragma unroll
  for (int m = 0; m < 16; ++m) {
    long long r = row0 + m;
    f32x4v v = {0.f, 0.f, 0.f, 0.f};
    if (r < NN) v = __builtin_nontemporal_load((const f32x4v*)(x + r * 128 + lane * 4));
    unsigned int lo = (unsigned int)f2bf(v.x) | ((unsigned int)f2bf(v.y) << 16);
    unsigned int hi = (unsigned int)f2bf(v.z) | ((unsigned int)f2bf(v.w) << 16);
    *(uint2*)&at[m * 128 + lane * 4] = make_uint2(lo, hi);
  }
  WAVE_WAIT_DS();

  f32x8 acc[4] = {};
  #pragma unroll
  for (int k0 = 0; k0 < 128; k0 += 32) {
    bf16x16 a = load_a(at, 128, lane, k0);
    #pragma unroll
    for (int c = 0; c < 4; ++c) {
      bf16x16 b = load_b(wt, 128, lane, c, k0);
      acc[c] = wmma_bf16(a, b, acc[c]);
    }
  }

  int n = lane & 15, hi = lane >> 4;
  #pragma unroll
  for (int c = 0; c < 4; ++c) {
    int col = c * 16 + n;
    float bb = bsh[col];
    #pragma unroll
    for (int r = 0; r < 8; ++r) {
      long long row = row0 + hi * 8 + r;
      if (row < NN) h[row * 64 + col] = lrelu(acc[c][r] + bb);
    }
  }
}

// ---------------------------------------------------------------------------
// agg zero-fill (agg should become L2-resident -> regular stores)
// ---------------------------------------------------------------------------
__global__ void __launch_bounds__(256)
zero_kernel(float* __restrict__ p, long long nelem) {
  long long i = (long long)blockIdx.x * 256 + threadIdx.x;
  long long stride = (long long)gridDim.x * 256;
  for (; i < nelem; i += stride) p[i] = 0.f;
}

// ---------------------------------------------------------------------------
// message + scatter: agg[dst] += relu(h[src] + e)   (4 threads per edge)
// e stream NT; h gather + agg atomics hit L2 (both arrays fit in 192MB L2)
// ---------------------------------------------------------------------------
__global__ void __launch_bounds__(256)
message_kernel(const float* __restrict__ h, const unsigned short* __restrict__ e,
               const int* __restrict__ src, const int* __restrict__ dst,
               float* __restrict__ agg) {
  long long t = (long long)blockIdx.x * 256 + threadIdx.x;
  long long eidx = t >> 2;
  int q = (int)(t & 3);
  if (eidx >= NE) return;

  long long s = src[eidx];
  long long d = dst[eidx];

  const float4* hp = (const float4*)(h + s * 64 + q * 16);
  const u32x4v* ep = (const u32x4v*)(e + eidx * 64 + q * 16);
  float* ap = agg + d * 64 + q * 16;

  u32x4v e0 = __builtin_nontemporal_load(ep);
  u32x4v e1 = __builtin_nontemporal_load(ep + 1);
  unsigned int eu[8] = {e0.x, e0.y, e0.z, e0.w, e1.x, e1.y, e1.z, e1.w};
  float ev[16];
  #pragma unroll
  for (int i = 0; i < 8; ++i) {
    ev[2 * i]     = bf2f((unsigned short)(eu[i] & 0xffffu));
    ev[2 * i + 1] = bf2f((unsigned short)(eu[i] >> 16));
  }
  #pragma unroll
  for (int i = 0; i < 4; ++i) {
    float4 hv = hp[i];
    float m0 = hv.x + ev[4 * i + 0]; m0 = m0 > 0.f ? m0 : 0.f;
    float m1 = hv.y + ev[4 * i + 1]; m1 = m1 > 0.f ? m1 : 0.f;
    float m2 = hv.z + ev[4 * i + 2]; m2 = m2 > 0.f ? m2 : 0.f;
    float m3 = hv.w + ev[4 * i + 3]; m3 = m3 > 0.f ? m3 : 0.f;
    if (m0 != 0.f) unsafeAtomicAdd(ap + 4 * i + 0, m0);
    if (m1 != 0.f) unsafeAtomicAdd(ap + 4 * i + 1, m1);
    if (m2 != 0.f) unsafeAtomicAdd(ap + 4 * i + 2, m2);
    if (m3 != 0.f) unsafeAtomicAdd(ap + 4 * i + 3, m3);
  }
}

// ---------------------------------------------------------------------------
// hout = lrelu( lrelu((h+agg) @ W1 + b1) @ W2 + b2 )   (fused 2-GEMM MLP)
// h/agg/hout are node state -> regular temporal (L2-resident)
// ---------------------------------------------------------------------------
__global__ void __launch_bounds__(256)
node_mlp_kernel(const float* __restrict__ h, const float* __restrict__ agg,
                const float* __restrict__ W1, const float* __restrict__ b1,
                const float* __restrict__ W2, const float* __restrict__ b2,
                float* __restrict__ hout) {
  __shared__ __align__(16) unsigned short wt1[64 * 64];
  __shared__ __align__(16) unsigned short wt2[64 * 64];
  __shared__ __align__(16) unsigned short astage[8][16 * 64];
  __shared__ float b1s[64], b2s[64];

  int tid = threadIdx.x;
  for (int i = tid; i < 64 * 64; i += 256) {
    int k = i >> 6, n = i & 63;
    wt1[n * 64 + k] = f2bf(W1[i]);
    wt2[n * 64 + k] = f2bf(W2[i]);
  }
  if (tid < 64) { b1s[tid] = b1[tid]; b2s[tid] = b2[tid]; }
  __syncthreads();

  int wave = tid >> 5, lane = tid & 31;
  long long row0 = (long long)blockIdx.x * 128 + wave * 16;
  unsigned short* at = astage[wave];

  // z = h + agg -> bf16 staging
  #pragma unroll
  for (int m = 0; m < 16; ++m) {
    long long r = row0 + m;
    float2 a = make_float2(0.f, 0.f), g = make_float2(0.f, 0.f);
    if (r < NN) {
      a = *(const float2*)(h + r * 64 + lane * 2);
      g = *(const float2*)(agg + r * 64 + lane * 2);
    }
    *(unsigned int*)&at[m * 64 + lane * 2] =
        (unsigned int)f2bf(a.x + g.x) | ((unsigned int)f2bf(a.y + g.y) << 16);
  }
  WAVE_WAIT_DS();

  f32x8 acc[4] = {};
  #pragma unroll
  for (int k0 = 0; k0 < 64; k0 += 32) {
    bf16x16 a = load_a(at, 64, lane, k0);
    #pragma unroll
    for (int c = 0; c < 4; ++c)
      acc[c] = wmma_bf16(a, load_b(wt1, 64, lane, c, k0), acc[c]);
  }

  // t = lrelu(acc + b1) -> bf16 staging (reuse wave-private region)
  int n = lane & 15, hi = lane >> 4;
  #pragma unroll
  for (int c = 0; c < 4; ++c) {
    int col = c * 16 + n;
    float bb = b1s[col];
    #pragma unroll
    for (int r = 0; r < 8; ++r)
      at[(hi * 8 + r) * 64 + col] = f2bf(lrelu(acc[c][r] + bb));
  }
  WAVE_WAIT_DS();

  f32x8 acc2[4] = {};
  #pragma unroll
  for (int k0 = 0; k0 < 64; k0 += 32) {
    bf16x16 a = load_a(at, 64, lane, k0);
    #pragma unroll
    for (int c = 0; c < 4; ++c)
      acc2[c] = wmma_bf16(a, load_b(wt2, 64, lane, c, k0), acc2[c]);
  }

  #pragma unroll
  for (int c = 0; c < 4; ++c) {
    int col = c * 16 + n;
    float bb = b2s[col];
    #pragma unroll
    for (int r = 0; r < 8; ++r) {
      long long row = row0 + hi * 8 + r;
      if (row < NN) hout[row * 64 + col] = lrelu(acc2[c][r] + bb);
    }
  }
}

// ---------------------------------------------------------------------------
// out[N,21] = h[N,64] @ W[64,21] + b   (WMMA, N-dim padded to 32)
// final output is write-once -> NT stores
// ---------------------------------------------------------------------------
__global__ void __launch_bounds__(256)
out_kernel(const float* __restrict__ h, const float* __restrict__ W,
           const float* __restrict__ bias, float* __restrict__ out) {
  __shared__ __align__(16) unsigned short wt[32 * 64];        // Wt[n][k], n>=21 zero
  __shared__ __align__(16) unsigned short astage[8][16 * 64];
  __shared__ float bsh[32];

  int tid = threadIdx.x;
  for (int i = tid; i < 32 * 64; i += 256) {
    int n = i >> 6, k = i & 63;
    wt[n * 64 + k] = (n < OD) ? f2bf(W[k * OD + n]) : (unsigned short)0;
  }
  if (tid < 32) bsh[tid] = (tid < OD) ? bias[tid] : 0.f;
  __syncthreads();

  int wave = tid >> 5, lane = tid & 31;
  long long row0 = (long long)blockIdx.x * 128 + wave * 16;
  unsigned short* at = astage[wave];

  #pragma unroll
  for (int m = 0; m < 16; ++m) {
    long long r = row0 + m;
    float2 v = make_float2(0.f, 0.f);
    if (r < NN) v = *(const float2*)(h + r * 64 + lane * 2);
    *(unsigned int*)&at[m * 64 + lane * 2] =
        (unsigned int)f2bf(v.x) | ((unsigned int)f2bf(v.y) << 16);
  }
  WAVE_WAIT_DS();

  f32x8 acc[2] = {};
  #pragma unroll
  for (int k0 = 0; k0 < 64; k0 += 32) {
    bf16x16 a = load_a(at, 64, lane, k0);
    #pragma unroll
    for (int c = 0; c < 2; ++c)
      acc[c] = wmma_bf16(a, load_b(wt, 64, lane, c, k0), acc[c]);
  }

  int n = lane & 15, hi = lane >> 4;
  #pragma unroll
  for (int c = 0; c < 2; ++c) {
    int col = c * 16 + n;
    if (col >= OD) continue;
    float bb = bsh[col];
    #pragma unroll
    for (int r = 0; r < 8; ++r) {
      long long row = row0 + hi * 8 + r;
      if (row < NN)
        __builtin_nontemporal_store(acc[c][r] + bb, out + row * OD + col);
    }
  }
}

// ---------------------------------------------------------------------------
extern "C" void kernel_launch(void* const* d_in, const int* in_sizes, int n_in,
                              void* d_out, int out_size, void* d_ws, size_t ws_size,
                              hipStream_t stream) {
  const float* x         = (const float*)d_in[0];
  const float* edge_attr = (const float*)d_in[1];
  const int*   edge_idx  = (const int*)d_in[2];     // [2, NE] (JAX default int32)
  const float* node_w    = (const float*)d_in[3];
  const float* node_b    = (const float*)d_in[4];
  const float* edge_w    = (const float*)d_in[5];
  const float* edge_b    = (const float*)d_in[6];
  const float* c1_w1     = (const float*)d_in[7];
  const float* c1_b1     = (const float*)d_in[8];
  const float* c1_w2     = (const float*)d_in[9];
  const float* c1_b2     = (const float*)d_in[10];
  const float* c2_w1     = (const float*)d_in[11];
  const float* c2_b1     = (const float*)d_in[12];
  const float* c2_w2     = (const float*)d_in[13];
  const float* c2_b2     = (const float*)d_in[14];
  const float* out_w     = (const float*)d_in[15];
  const float* out_b     = (const float*)d_in[16];
  float* out = (float*)d_out;

  // workspace layout
  char* ws = (char*)d_ws;
  unsigned short* e_bf = (unsigned short*)ws;                         // NE*64 bf16
  float* hA  = (float*)(ws + (size_t)NE * 64 * sizeof(unsigned short));
  float* hB  = hA + (size_t)NN * 64;
  float* agg = hB + (size_t)NN * 64;

  const int* src = edge_idx;
  const int* dst = edge_idx + NE;

  const dim3 blk(256);
  const int nodeBlocks = (NN + 127) / 128;   // 782
  const int edgeBlocks = (NE + 127) / 128;   // 12500
  const int msgBlocks  = (int)(((long long)NE * 4 + 255) / 256); // 25000
  const long long aggElems = (long long)NN * 64;

  node_linear_kernel<<<nodeBlocks, blk, 0, stream>>>(x, node_w, node_b, hA);
  edge_linear_kernel<<<edgeBlocks, blk, 0, stream>>>(edge_attr, edge_w, edge_b, e_bf);

  zero_kernel<<<2048, blk, 0, stream>>>(agg, aggElems);
  message_kernel<<<msgBlocks, blk, 0, stream>>>(hA, e_bf, src, dst, agg);
  node_mlp_kernel<<<nodeBlocks, blk, 0, stream>>>(hA, agg, c1_w1, c1_b1, c1_w2, c1_b2, hB);

  zero_kernel<<<2048, blk, 0, stream>>>(agg, aggElems);
  message_kernel<<<msgBlocks, blk, 0, stream>>>(hB, e_bf, src, dst, agg);
  node_mlp_kernel<<<nodeBlocks, blk, 0, stream>>>(hB, agg, c2_w1, c2_b1, c2_w2, c2_b2, hA);

  out_kernel<<<nodeBlocks, blk, 0, stream>>>(hA, out_w, out_b, out);
}
